// VQSign_54786602828158
// MI455X (gfx1250) — matrix operand order repaired and checked by reading, not compiled, not run on hardware
//
#include <hip/hip_runtime.h>
#include <hip/hip_bf16.h>
#include <math.h>

// ---------------------------------------------------------------- types
typedef __bf16 bf16_t;
typedef __attribute__((ext_vector_type(16))) bf16_t bf16x16;
typedef __attribute__((ext_vector_type(8)))  bf16_t bf16x8;
typedef __attribute__((ext_vector_type(8)))  float  f32x8;

#define Bsz 32
#define Tsz 512
#define Dsz 1024
#define KCB 256
#define Gsz 3072   // 3*Dsz

// ---------------------------------------------------------------- WMMA helpers
__device__ __forceinline__ f32x8 wmma_bf16(bf16x16 a, bf16x16 b, f32x8 c) {
  return __builtin_amdgcn_wmma_f32_16x16x32_bf16(
      /*neg_a=*/false, a, /*neg_b=*/false, b,
      /*c_mod=*/(short)0, c, /*reuse_a=*/false, /*reuse_b=*/false);
}

// A-matrix 16x32 bf16 fragment from row-major [rows x ld], tile at (row0, k0).
// lanes 0-15: M=ln, K = k0 + 0..15; lanes 16-31: K += 16.
__device__ __forceinline__ bf16x16 load_a_frag(const bf16_t* A, int row0, int ld,
                                               int k0, int ln, int hx) {
  const bf16_t* p = A + (size_t)(row0 + ln) * ld + k0 + hx * 16;
  return *(const bf16x16*)p;
}

// Same A fragment but from an fp32 tile (ld floats), converting at build time.
__device__ __forceinline__ bf16x16 load_a_frag_f32(const float* A, int ld,
                                                   int ln, int hx) {
  const float* p = A + ln * ld + hx * 16;
  bf16x16 f;
#pragma unroll
  for (int i = 0; i < 16; ++i) f[i] = (bf16_t)p[i];
  return f;
}

// B-matrix 32x16 bf16 fragment; W is row-major [N x ld] holding output-col rows
// (i.e. B[k][n] = W[col0+n][k0+k]).  lanes 0-15: K=k0+0..15; lanes 16-31: +16.
__device__ __forceinline__ bf16x16 load_b_frag(const bf16_t* W, int col0, int ld,
                                               int k0, int ln, int hx) {
  const bf16_t* p = W + (size_t)(col0 + ln) * ld + k0 + hx * 16;
  return *(const bf16x16*)p;
}

__device__ __forceinline__ float wave_sum(float v) {
#pragma unroll
  for (int m = 16; m; m >>= 1) v += __shfl_xor(v, m, 32);
  return v;
}
__device__ __forceinline__ float sigmoidf_(float x) { return 1.0f / (1.0f + expf(-x)); }

// ---- CDNA5 async global->LDS copy (ASYNCcnt path), 16B per lane ----
__device__ __forceinline__ void async_copy_b128(unsigned lds_addr, const void* gaddr) {
  asm volatile("global_load_async_to_lds_b128 %0, %1, off"
               :: "v"(lds_addr), "v"(gaddr) : "memory");
}
__device__ __forceinline__ void wait_asynccnt0() {
  asm volatile("s_wait_asynccnt 0x0" ::: "memory");
}

// ---------------------------------------------------------------- convert fp32 -> bf16
__global__ void cvt_bf16_kernel(const float* __restrict__ src, bf16_t* __restrict__ dst, int n) {
  for (int i = blockIdx.x * blockDim.x + threadIdx.x; i < n; i += gridDim.x * blockDim.x)
    dst[i] = (bf16_t)src[i];
}

// codebook row norms: one wave per row
__global__ void cbnorm_kernel(const float* __restrict__ cb, float* __restrict__ nrm) {
  int row = blockIdx.x;
  int lane = threadIdx.x;
  float s = 0.f;
  for (int c = lane; c < Dsz; c += 32) {
    float v = cb[(size_t)row * Dsz + c];
    s += v * v;
  }
  s = wave_sum(s);
  if (lane == 0) nrm[row] = s;
}

// ---------------------------------------------------------------- quantize
// One block (128 thr / 4 waves) per 16 feature rows; wave w owns cols [64w,64w+64).
// Feature tile is double-buffered in LDS via async global->LDS DMA; the next
// K-tile's copy overlaps the current tile's 4 WMMAs.
__global__ __launch_bounds__(128) void quantize_kernel(
    const float* __restrict__ feats, const float* __restrict__ cb,
    const bf16_t* __restrict__ cbB, const float* __restrict__ cbn,
    float* __restrict__ qOut, bf16_t* __restrict__ qB,
    float* __restrict__ idxOut, float* __restrict__ accSq) {
  __shared__ float aStage[2][16 * 32];   // fp32 tiles, 2 KB each
  __shared__ float d2[16 * KCB];
  __shared__ int bestIdx[16];
  __shared__ float wsum[4];

  const int tid = threadIdx.x;
  const int w = tid >> 5, lane = tid & 31, ln = lane & 15, hx = lane >> 4;
  const int row0 = blockIdx.x * 16;

  // this thread's 16-byte slice of the 16x32 fp32 tile
  const int sr = tid >> 3;            // row 0..15
  const int sc = (tid & 7) * 4;       // float col 0,4,..,28
  const float* gRow = feats + (size_t)(row0 + sr) * Dsz + sc;

  f32x8 acc[4];
#pragma unroll
  for (int i = 0; i < 4; ++i) acc[i] = (f32x8)0.0f;

  // prologue: stage k-tile 0 into buffer 0
  async_copy_b128((unsigned)(size_t)&aStage[0][sr * 32 + sc], (const void*)gRow);
  wait_asynccnt0();
  __syncthreads();

  for (int kb = 0; kb < Dsz / 32; ++kb) {
    const int cur = kb & 1;
    const int k0 = kb * 32;
    bf16x16 aF = load_a_frag_f32(aStage[cur], 32, ln, hx);
    if (kb + 1 < Dsz / 32)  // kick DMA for next tile into the other buffer
      async_copy_b128((unsigned)(size_t)&aStage[cur ^ 1][sr * 32 + sc],
                      (const void*)(gRow + k0 + 32));
#pragma unroll
    for (int i = 0; i < 4; ++i) {
      bf16x16 bF = load_b_frag(cbB, w * 64 + i * 16, Dsz, k0, ln, hx);
      acc[i] = wmma_bf16(aF, bF, acc[i]);
    }
    wait_asynccnt0();
    __syncthreads();
  }
  // d2 = |c|^2 - 2 f.c   (|f|^2 irrelevant to argmin)
#pragma unroll
  for (int i = 0; i < 4; ++i) {
    int col = w * 64 + i * 16 + ln;
    float cn = cbn[col];
#pragma unroll
    for (int rr = 0; rr < 8; ++rr) {
      int row = rr + hx * 8;
      d2[row * KCB + col] = cn - 2.0f * acc[i][rr];
    }
  }
  __syncthreads();
  if (tid < 16) {
    float best = d2[tid * KCB];
    int bi = 0;
    for (int c = 1; c < KCB; ++c) {
      float v = d2[tid * KCB + c];
      if (v < best) { best = v; bi = c; }
    }
    bestIdx[tid] = bi;
    idxOut[row0 + tid] = (float)bi;
  }
  __syncthreads();
  // gather codebook rows + VQ squared error
  float sq = 0.f;
  for (int e = tid; e < 16 * Dsz; e += 128) {
    int r = e >> 10, c = e & (Dsz - 1);
    int bi = bestIdx[r];
    float q = cb[(size_t)bi * Dsz + c];
    float f = feats[(size_t)(row0 + r) * Dsz + c];
    float d = f - q;
    sq += d * d;
    qOut[(size_t)(row0 + r) * Dsz + c] = q;
    qB[(size_t)(row0 + r) * Dsz + c] = (bf16_t)q;
  }
  sq = wave_sum(sq);
  if (lane == 0) wsum[w] = sq;
  __syncthreads();
  if (tid == 0) atomicAdd(accSq, wsum[0] + wsum[1] + wsum[2] + wsum[3]);
}

// ---------------------------------------------------------------- generic GEMM
// out[MxN] = A[MxK](bf16) @ W[NxK]^T(bf16) + bias[N].  Block = 4 waves,
// wave w -> rows [bm*64 + 16w, +16), cols [bn*64, +64).
__global__ __launch_bounds__(128) void gemm_bias_kernel(
    const bf16_t* __restrict__ A, const bf16_t* __restrict__ W,
    const float* __restrict__ bias, float* __restrict__ out, int N, int K) {
  const int tid = threadIdx.x;
  const int w = tid >> 5, lane = tid & 31, ln = lane & 15, hx = lane >> 4;
  const int m0 = blockIdx.x * 64 + w * 16;
  const int c0 = blockIdx.y * 64;

  f32x8 acc[4];
#pragma unroll
  for (int i = 0; i < 4; ++i) acc[i] = (f32x8)0.0f;

  for (int k0 = 0; k0 < K; k0 += 32) {
    if (k0 + 32 < K) {  // prefetch next k-block lines (global_prefetch_b8)
      __builtin_prefetch(A + (size_t)(m0 + ln) * K + k0 + 32, 0, 1);
      __builtin_prefetch(W + (size_t)(c0 + ln) * K + k0 + 32, 0, 1);
    }
    bf16x16 aF = load_a_frag(A, m0, K, k0, ln, hx);
#pragma unroll
    for (int i = 0; i < 4; ++i) {
      bf16x16 bF = load_b_frag(W, c0 + i * 16, K, k0, ln, hx);
      acc[i] = wmma_bf16(aF, bF, acc[i]);
    }
  }
#pragma unroll
  for (int i = 0; i < 4; ++i) {
    int col = c0 + i * 16 + ln;
    float bv = bias ? bias[col] : 0.0f;
#pragma unroll
    for (int rr = 0; rr < 8; ++rr)
      out[(size_t)(m0 + rr + hx * 8) * N + col] = acc[i][rr] + bv;
  }
}

// ---------------------------------------------------------------- GRU recurrence
// Single 512-thread workgroup (16 waves). Wave w owns j-tiles {w+16*jj},
// jj=0..3, for both m-tiles (rows b 0..15 / 16..31).  h kept in registers,
// bf16 mirror in LDS feeds WMMA A-fragments.  W_hh stays resident in L2.
__global__ __launch_bounds__(512) void gru_kernel(
    const float* __restrict__ gi, const bf16_t* __restrict__ WhhB,
    const float* __restrict__ bhh, bf16_t* __restrict__ hsB) {
  __shared__ bf16_t hS[Bsz * Dsz];  // 64 KB

  const int tid = threadIdx.x;
  const int w = tid >> 5, lane = tid & 31, ln = lane & 15, hx = lane >> 4;

  for (int e = tid; e < Bsz * Dsz; e += 512) hS[e] = (bf16_t)0.0f;

  f32x8 hreg[8];  // [m*4 + jj]
#pragma unroll
  for (int i = 0; i < 8; ++i) hreg[i] = (f32x8)0.0f;

  float bR[4], bZ[4], bN[4];
#pragma unroll
  for (int jj = 0; jj < 4; ++jj) {
    int j0 = (w + 16 * jj) * 16 + ln;
    bR[jj] = bhh[j0];
    bZ[jj] = bhh[Dsz + j0];
    bN[jj] = bhh[2 * Dsz + j0];
  }
  __syncthreads();

  for (int t = 0; t < Tsz; ++t) {
#pragma unroll 1
    for (int jj = 0; jj < 4; ++jj) {
      const int jt = w + 16 * jj;
      f32x8 aR0 = (f32x8)0.0f, aR1 = (f32x8)0.0f;
      f32x8 aZ0 = (f32x8)0.0f, aZ1 = (f32x8)0.0f;
      f32x8 aN0 = (f32x8)0.0f, aN1 = (f32x8)0.0f;
#pragma unroll 1
      for (int kb = 0; kb < Dsz / 32; ++kb) {
        const int k0 = kb * 32;
        bf16x16 a0 = load_a_frag(hS, 0, Dsz, k0, ln, hx);
        bf16x16 a1 = load_a_frag(hS, 16, Dsz, k0, ln, hx);
        bf16x16 bF;
        bF = load_b_frag(WhhB, 0 * Dsz + jt * 16, Dsz, k0, ln, hx);
        aR0 = wmma_bf16(a0, bF, aR0); aR1 = wmma_bf16(a1, bF, aR1);
        bF = load_b_frag(WhhB, 1 * Dsz + jt * 16, Dsz, k0, ln, hx);
        aZ0 = wmma_bf16(a0, bF, aZ0); aZ1 = wmma_bf16(a1, bF, aZ1);
        bF = load_b_frag(WhhB, 2 * Dsz + jt * 16, Dsz, k0, ln, hx);
        aN0 = wmma_bf16(a0, bF, aN0); aN1 = wmma_bf16(a1, bF, aN1);
      }
      const int j = jt * 16 + ln;
#pragma unroll
      for (int m = 0; m < 2; ++m) {
        f32x8 hr_ = (m == 0) ? aR0 : aR1;
        f32x8 hz_ = (m == 0) ? aZ0 : aZ1;
        f32x8 hn_ = (m == 0) ? aN0 : aN1;
#pragma unroll
        for (int rr = 0; rr < 8; ++rr) {
          int b = m * 16 + rr + hx * 8;
          size_t gbase = ((size_t)b * Tsz + t) * Gsz;
          float ir = gi[gbase + j];
          float iz = gi[gbase + Dsz + j];
          float in_ = gi[gbase + 2 * Dsz + j];
          float rg = sigmoidf_(ir + hr_[rr] + bR[jj]);
          float zg = sigmoidf_(iz + hz_[rr] + bZ[jj]);
          float ng = tanhf(in_ + rg * (hn_[rr] + bN[jj]));
          float hp = hreg[m * 4 + jj][rr];
          hreg[m * 4 + jj][rr] = (1.0f - zg) * ng + zg * hp;
        }
      }
    }
    __syncthreads();  // everyone done reading old hS
#pragma unroll
    for (int jj = 0; jj < 4; ++jj) {
      const int j = (w + 16 * jj) * 16 + ln;
#pragma unroll
      for (int m = 0; m < 2; ++m) {
#pragma unroll
        for (int rr = 0; rr < 8; ++rr) {
          int b = m * 16 + rr + hx * 8;
          bf16_t hv = (bf16_t)hreg[m * 4 + jj][rr];
          hS[b * Dsz + j] = hv;
          hsB[((size_t)b * Tsz + t) * Dsz + j] = hv;
        }
      }
    }
    __syncthreads();  // hS(t) published
  }
}

// ---------------------------------------------------------------- CP loss: positive term
__global__ void pos_loss_kernel(const float* __restrict__ ctx,
                                const float* __restrict__ feats,
                                float* __restrict__ accPos) {
  const int kidx = blockIdx.y, k = kidx + 1, L = Tsz - k;
  const int gw = blockIdx.x * (blockDim.x >> 5) + (threadIdx.x >> 5);
  const int lane = threadIdx.x & 31;
  if (gw >= Bsz * L) return;
  const int b = gw / L, l = gw % L;
  const float* h = ctx + ((size_t)b * Tsz + l) * Dsz;
  const float* p = feats + ((size_t)b * Tsz + l + k) * Dsz;
  float s = 0.f;
  for (int c = lane; c < Dsz; c += 32) s += h[c] * p[c];
  s = wave_sum(s);
  if (lane == 0) {
    float loss = -logf(sigmoidf_(s) + 1e-8f);
    atomicAdd(&accPos[b], loss);
  }
}

// ---------------------------------------------------------------- CP loss: negative term
// One block (4 waves) per (k, l): 32x32 logits via WMMA, fp32->bf16 on the fly.
__global__ __launch_bounds__(128) void neg_loss_kernel(
    const float* __restrict__ ctx, const float* __restrict__ feats,
    const int* __restrict__ perm, float* __restrict__ accNeg) {
  const int kidx = blockIdx.y, k = kidx + 1;
  const int lt = blockIdx.x;
  if (lt >= Tsz - k) return;
  const int tid = threadIdx.x;
  const int w = tid >> 5, lane = tid & 31, ln = lane & 15, hx = lane >> 4;
  const int mt = w >> 1, nt = w & 1;
  const int pr = perm[kidx * Bsz + nt * 16 + ln];

  f32x8 acc = (f32x8)0.0f;
  for (int k0 = 0; k0 < Dsz; k0 += 32) {
    // A: h rows (b), contiguous K (16 per lane half)
    const float* ab = ctx + ((size_t)(mt * 16 + ln) * Tsz + lt) * Dsz + k0 + hx * 16;
    bf16x16 aF, bF;
#pragma unroll
    for (int i = 0; i < 16; ++i) aF[i] = (bf16_t)ab[i];
    const float* bb = feats + ((size_t)pr * Tsz + lt) * Dsz + k0 + hx * 16;
#pragma unroll
    for (int i = 0; i < 16; ++i) bF[i] = (bf16_t)bb[i];
    acc = wmma_bf16(aF, bF, acc);
  }
  float s = 0.f;
#pragma unroll
  for (int rr = 0; rr < 8; ++rr) {
    float logit = acc[rr];
    // 1 - sigmoid(x) == sigmoid(-x)
    s += -logf(sigmoidf_(-logit) + 1e-8f);
  }
  s = wave_sum(s);
  if (lane == 0) atomicAdd(accNeg, s);
}

// ---------------------------------------------------------------- finalize
__global__ void finalize_kernel(const float* __restrict__ accPos,
                                const float* __restrict__ accNeg,
                                const float* __restrict__ accSq,
                                float* __restrict__ lossOut) {
  int t = threadIdx.x;
  if (t < Bsz) {
    float cp = (accPos[t] + 0.25f * accNeg[0] * (1.0f / 1024.0f)) * (1.0f / 1527.0f);
    float vq = 1.25f * accSq[0] * (1.0f / 16777216.0f);
    lossOut[t] = cp + vq;
  }
}

// ---------------------------------------------------------------- host launcher
extern "C" void kernel_launch(void* const* d_in, const int* in_sizes, int n_in,
                              void* d_out, int out_size, void* d_ws, size_t ws_size,
                              hipStream_t stream) {
  const float* feats  = (const float*)d_in[0];
  const float* cb     = (const float*)d_in[1];
  const float* Wih    = (const float*)d_in[2];
  const float* Whh    = (const float*)d_in[3];
  const float* bih    = (const float*)d_in[4];
  const float* bhh    = (const float*)d_in[5];
  const float* Wproj  = (const float*)d_in[6];
  const float* bproj  = (const float*)d_in[7];
  const int*   perm   = (const int*)d_in[8];

  float* outQ    = (float*)d_out;                       // 32*512*1024
  float* outIdx  = outQ + (size_t)Bsz * Tsz * Dsz;      // 32*512
  float* outLoss = outIdx + Bsz * Tsz;                  // 32

  // workspace carve-up (256B aligned)
  unsigned char* p = (unsigned char*)d_ws;
  auto carve = [&](size_t bytes) {
    void* r = (void*)p;
    p += (bytes + 255) & ~(size_t)255;
    return r;
  };
  bf16_t* quantB = (bf16_t*)carve((size_t)Bsz * Tsz * Dsz * 2);
  bf16_t* hsB    = (bf16_t*)carve((size_t)Bsz * Tsz * Dsz * 2);
  float*  gi     = (float*) carve((size_t)Bsz * Tsz * Gsz * 4);
  float*  ctx    = (float*) carve((size_t)Bsz * Tsz * Dsz * 4);
  bf16_t* cbB    = (bf16_t*)carve((size_t)KCB * Dsz * 2);
  bf16_t* WihB   = (bf16_t*)carve((size_t)Gsz * Dsz * 2);
  bf16_t* WhhB   = (bf16_t*)carve((size_t)Gsz * Dsz * 2);
  bf16_t* WprojB = (bf16_t*)carve((size_t)Dsz * Dsz * 2);
  float*  cbn    = (float*) carve(KCB * 4);
  float*  acc    = (float*) carve(64 * 4);   // [0..31]=pos, [32]=neg, [33]=sq
  float* accPos = acc;
  float* accNeg = acc + 32;
  float* accSq  = acc + 33;

  hipMemsetAsync(acc, 0, 64 * 4, stream);

  // weight / codebook conversions
  cvt_bf16_kernel<<<1024, 256, 0, stream>>>(cb, cbB, KCB * Dsz);
  cvt_bf16_kernel<<<4096, 256, 0, stream>>>(Wih, WihB, Gsz * Dsz);
  cvt_bf16_kernel<<<4096, 256, 0, stream>>>(Whh, WhhB, Gsz * Dsz);
  cvt_bf16_kernel<<<2048, 256, 0, stream>>>(Wproj, WprojB, Dsz * Dsz);
  cbnorm_kernel<<<KCB, 32, 0, stream>>>(cb, cbn);

  // 1) vector quantization (+ VQ loss accumulation), async-LDS staged
  quantize_kernel<<<(Bsz * Tsz) / 16, 128, 0, stream>>>(
      feats, cb, cbB, cbn, outQ, quantB, outIdx, accSq);

  // 2) gi = quantized @ W_ih^T + b_ih   (big parallel GEMM)
  gemm_bias_kernel<<<dim3((Bsz * Tsz) / 64, Gsz / 64), 128, 0, stream>>>(
      quantB, WihB, bih, gi, Gsz, Dsz);

  // 3) GRU recurrence (sequential, single persistent workgroup)
  gru_kernel<<<1, 512, 0, stream>>>(gi, WhhB, bhh, hsB);

  // 4) context_proj = hidden @ W_proj^T + b_proj
  gemm_bias_kernel<<<dim3((Bsz * Tsz) / 64, Dsz / 64), 128, 0, stream>>>(
      hsB, WprojB, bproj, ctx, Dsz, Dsz);

  // 5) CP loss
  pos_loss_kernel<<<dim3((Bsz * (Tsz - 1) + 7) / 8, 3), 256, 0, stream>>>(ctx, feats, accPos);
  neg_loss_kernel<<<dim3(Tsz - 1, 3), 128, 0, stream>>>(ctx, feats, perm, accNeg);

  // 6) total loss per batch element
  finalize_kernel<<<1, 32, 0, stream>>>(accPos, accNeg, accSq, outLoss);
}
